// Encoder_PointnetPlusPlus_56453050138937
// MI455X (gfx1250) — matrix operand order repaired and checked by reading, not compiled, not run on hardware
//
#include <hip/hip_runtime.h>
#include <hip/hip_bf16.h>

// ---------------------------------------------------------------------------
// PointNet++ encoder for MI455X (gfx1250, wave32):
//  - WMMA 16x16x32 f16->f32 for all GEMMs
//  - GLOBAL_LOAD_ASYNC_TO_LDS_B128 (ASYNCcnt) for A-tile staging
//  - TENSOR_LOAD_TO_LDS (TDM, TENSORcnt) for W-tile staging in the MLP head
// ---------------------------------------------------------------------------

#define B_    16
#define N_    8192
#define NP1_  512
#define NP2_  128
#define K_    50
#define R1SQ  0.04f
#define R2SQ  0.16f

#define USE_ASYNC 1
#if __has_builtin(__builtin_amdgcn_tensor_load_to_lds) && __has_builtin(__builtin_amdgcn_s_wait_tensorcnt)
#define HAS_TDM 1
#else
#define HAS_TDM 0
#endif

typedef _Float16 v16h __attribute__((ext_vector_type(16)));
typedef float    v8f  __attribute__((ext_vector_type(8)));
typedef unsigned int v4u  __attribute__((ext_vector_type(4)));
typedef int          v8i_ __attribute__((ext_vector_type(8)));
typedef int          v4i_ __attribute__((ext_vector_type(4)));

static __device__ __forceinline__ v8f zero8() {
    v8f z = {0.f,0.f,0.f,0.f,0.f,0.f,0.f,0.f};
    return z;
}

// LDS byte offset of a generic pointer to __shared__ data: flat LDS aperture
// keeps the LDS address in addr[31:0] (ISA 10.2 aperture mapping).
static __device__ __forceinline__ unsigned lds_off(const void* p) {
    return (unsigned)(size_t)p;
}

// 16-byte async copy global -> LDS (ASYNCcnt-tracked, no VGPR data path).
static __device__ __forceinline__ void async_copy16(unsigned lds, const void* g) {
    asm volatile("global_load_async_to_lds_b128 %0, %1, off"
                 :: "v"(lds), "v"(g) : "memory");
}
static __device__ __forceinline__ void async_wait0() {
    asm volatile("s_wait_asynccnt 0x0" ::: "memory");
}

#if HAS_TDM
// One-descriptor TDM load of a dense 2D f16 tile (rows x cols, row stride
// `strideElems` in global) into LDS at byte offset `ldsAddr` (packed rows).
// D# packing per CDNA5 ISA 8.3/8.4 (groups 2/3 zero: 2-D tensor).
static __device__ __forceinline__ void tdm_load_tile_f16(
    unsigned ldsAddr, const void* gptr, unsigned rows, unsigned cols,
    unsigned strideElems)
{
    const unsigned long long ga = (unsigned long long)(size_t)gptr;
    v4u g0;
    g0.x = 1u;                                    // count=1, user mode
    g0.y = ldsAddr;                               // lds_addr
    g0.z = (unsigned)(ga & 0xffffffffu);          // global_addr[31:0]
    g0.w = (unsigned)((ga >> 32) & 0x01ffffffu)   // global_addr[56:32]
         | (2u << 30);                            // type=2 ("image")
    v8i_ g1;
    g1[0] = (int)(1u << 16);                      // data_size=2 bytes
    g1[1] = (int)((cols & 0xffffu) << 16);        // tensor_dim0[15:0]
    g1[2] = (int)(((cols >> 16) & 0xffffu) | ((rows & 0xffffu) << 16)); // dim0 hi | dim1 lo
    g1[3] = (int)(((rows >> 16) & 0xffffu) | ((cols & 0xffffu) << 16)); // dim1 hi | tile_dim0
    g1[4] = (int)(rows & 0xffffu);                // tile_dim1 (tile_dim2=0)
    g1[5] = (int)strideElems;                     // tensor_dim0_stride[31:0]
    g1[6] = 0;                                    // stride0 hi | stride1 lo
    g1[7] = 0;                                    // stride1 hi
    v4i_ gz = {0, 0, 0, 0};
#if __clang_major__ >= 23
    v8i_ gz8 = {0, 0, 0, 0, 0, 0, 0, 0};
    __builtin_amdgcn_tensor_load_to_lds(g0, g1, gz, gz, gz8, 0);
#else
    __builtin_amdgcn_tensor_load_to_lds(g0, g1, gz, gz, 0);
#endif
}
#endif

// ---------------------------------------------------------------------------
// Generic wave-level GEMM on LDS activations + global f16 weights.
// Either writes f16 to LDS `O` with bias+relu, or fused per-column max into
// LDS int array `smaxi` restricted to rows < maxRows.
// Fragment layouts per CDNA5 ISA 7.12.2 (wave32).
// ---------------------------------------------------------------------------
static __device__ __forceinline__ void wave_gemm(
    const _Float16* __restrict__ A, int as,
    const _Float16* __restrict__ W, int wn,
    const float*    __restrict__ bias,
    _Float16* O, int os,
    int* smaxi, int maxRows,
    int Mtiles, int Ntiles, int Ksteps, int tid)
{
    const int wave = tid >> 5;
    const int lane = tid & 31;
    const int l15  = lane & 15;
    const int h    = lane >> 4;

    for (int tile = wave; tile < Mtiles * Ntiles; tile += 4) {
        const int mB = (tile / Ntiles) * 16;
        const int nB = (tile % Ntiles) * 16;
        v8f acc = zero8();
        for (int ks = 0; ks < Ksteps; ++ks) {
            const int kb = ks * 32;
            v16h af, bf;
#pragma unroll
            for (int v = 0; v < 8; ++v) {
                int ka = kb + ((v < 4) ? (8 * h + 2 * v) : (16 + 8 * h + 2 * (v - 4)));
                af[2 * v]     = A[(mB + l15) * as + ka];
                af[2 * v + 1] = A[(mB + l15) * as + ka + 1];
                int k2 = kb + ((v < 4) ? (8 * h + 2 * v) : (16 + 8 * h + 2 * (v - 4)));
                bf[2 * v]     = W[(size_t)k2 * wn + nB + l15];
                bf[2 * v + 1] = W[(size_t)(k2 + 1) * wn + nB + l15];
            }
            acc = __builtin_amdgcn_wmma_f32_16x16x32_f16(
                false, af, false, bf, (short)0, acc, false, false);
        }
        const int ncol = nB + l15;
        const float bv = bias[ncol];
#pragma unroll
        for (int r = 0; r < 8; ++r) {
            const int Mr = mB + r + 8 * h;
            float val = fmaxf(acc[r] + bv, 0.f);
            if (O) {
                O[Mr * os + ncol] = (_Float16)val;
            } else if (Mr < maxRows) {
                atomicMax(&smaxi[ncol], __float_as_int(val));
            }
        }
    }
}

// ---------------------------------------------------------------------------
// FPS stage 1: B blocks x 1024 threads, 8 points/thread in registers.
// ---------------------------------------------------------------------------
__global__ __launch_bounds__(1024) void fps1_kernel(const float* __restrict__ points,
                                                    int* __restrict__ outIdx)
{
    __shared__ float sv[32];
    __shared__ int   si[32];
    __shared__ int   s_far;
    __shared__ float sc[3];
    const int b   = blockIdx.x;
    const int tid = threadIdx.x;
    const float* P = points + (size_t)b * N_ * 3;

    float dist[8];
#pragma unroll
    for (int j = 0; j < 8; ++j) dist[j] = 1e10f;
    int far = 0;

    for (int it = 0; it < NP1_; ++it) {
        if (tid == 0) {
            outIdx[b * NP1_ + it] = far;
            sc[0] = P[far * 3 + 0];
            sc[1] = P[far * 3 + 1];
            sc[2] = P[far * 3 + 2];
        }
        __syncthreads();
        const float cx = sc[0], cy = sc[1], cz = sc[2];
        float best = -1.f;
        int   bi   = 0;
#pragma unroll
        for (int j = 0; j < 8; ++j) {
            const int i = tid + 1024 * j;
            const float dx = P[i * 3 + 0] - cx;
            const float dy = P[i * 3 + 1] - cy;
            const float dz = P[i * 3 + 2] - cz;
            const float d  = dx * dx + dy * dy + dz * dz;
            dist[j] = fminf(dist[j], d);
            if (dist[j] > best) { best = dist[j]; bi = i; }
        }
        for (int off = 16; off; off >>= 1) {
            const float ov = __shfl_xor(best, off, 32);
            const int   oi = __shfl_xor(bi, off, 32);
            if (ov > best) { best = ov; bi = oi; }
        }
        if ((tid & 31) == 0) { sv[tid >> 5] = best; si[tid >> 5] = bi; }
        __syncthreads();
        if (tid < 32) {
            best = sv[tid]; bi = si[tid];
            for (int off = 16; off; off >>= 1) {
                const float ov = __shfl_xor(best, off, 32);
                const int   oi = __shfl_xor(bi, off, 32);
                if (ov > best) { best = ov; bi = oi; }
            }
            if (tid == 0) s_far = bi;
        }
        __syncthreads();
        far = s_far;
    }
}

// ---------------------------------------------------------------------------
// FPS stage 2: on 128-d features, B blocks x 512 threads, 1 point/thread.
// ---------------------------------------------------------------------------
__global__ __launch_bounds__(512) void fps2_kernel(const float* __restrict__ pts,
                                                   int* __restrict__ outIdx)
{
    __shared__ float sc[128];
    __shared__ float sv[16];
    __shared__ int   si[16];
    __shared__ int   s_far;
    const int b   = blockIdx.x;
    const int tid = threadIdx.x;
    const float* P = pts + (size_t)b * NP1_ * 128;

    float dist = 1e10f;
    int far = 0;
    for (int it = 0; it < NP2_; ++it) {
        if (tid == 0) outIdx[b * NP2_ + it] = far;
        if (tid < 128) sc[tid] = P[(size_t)far * 128 + tid];
        __syncthreads();
        float d = 0.f;
        for (int k = 0; k < 128; ++k) {
            const float df = P[(size_t)tid * 128 + k] - sc[k];
            d += df * df;
        }
        dist = fminf(dist, d);
        float best = dist; int bi = tid;
        for (int off = 16; off; off >>= 1) {
            const float ov = __shfl_xor(best, off, 32);
            const int   oi = __shfl_xor(bi, off, 32);
            if (ov > best) { best = ov; bi = oi; }
        }
        if ((tid & 31) == 0) { sv[tid >> 5] = best; si[tid >> 5] = bi; }
        __syncthreads();
        if (tid < 32) {
            best = (tid < 16) ? sv[tid] : -2.f;
            bi   = (tid < 16) ? si[tid] : 0;
            for (int off = 16; off; off >>= 1) {
                const float ov = __shfl_xor(best, off, 32);
                const int   oi = __shfl_xor(bi, off, 32);
                if (ov > best) { best = ov; bi = oi; }
            }
            if (tid == 0) s_far = bi;
        }
        __syncthreads();
        far = s_far;
    }
}

// ---------------------------------------------------------------------------
// Ball queries: first-K-by-index within radius, pad with n-1 (ref semantics).
// ---------------------------------------------------------------------------
__global__ void ballq1_kernel(const float* __restrict__ points,
                              const int* __restrict__ cidx,
                              int* __restrict__ idx1)
{
    const int gid = blockIdx.x * blockDim.x + threadIdx.x;
    if (gid >= B_ * NP1_) return;
    const int b  = gid >> 9;
    const int ci = cidx[gid];
    const float* cp = points + (size_t)(b * N_ + ci) * 3;
    const float cx = cp[0], cy = cp[1], cz = cp[2];
    int cnt = 0;
    const int base = gid * K_;
    for (int i = 0; i < N_ && cnt < K_; ++i) {
        const float* p = points + (size_t)(b * N_ + i) * 3;
        const float dx = p[0] - cx, dy = p[1] - cy, dz = p[2] - cz;
        if (dx * dx + dy * dy + dz * dz < R1SQ) idx1[base + cnt++] = i;
    }
    for (; cnt < K_; ++cnt) idx1[base + cnt] = N_ - 1;
}

__global__ void ballq2_kernel(const float* __restrict__ pts,
                              const int* __restrict__ cidx,
                              int* __restrict__ idx2)
{
    const int gid = blockIdx.x * blockDim.x + threadIdx.x;
    if (gid >= B_ * NP2_) return;
    const int b  = gid >> 7;
    const int ci = cidx[gid];
    const float* cp = pts + (size_t)(b * NP1_ + ci) * 128;
    int cnt = 0;
    const int base = gid * K_;
    for (int i = 0; i < NP1_ && cnt < K_; ++i) {
        const float* p = pts + (size_t)(b * NP1_ + i) * 128;
        float d = 0.f;
        for (int k = 0; k < 128; ++k) {
            const float df = p[k] - cp[k];
            d += df * df;
        }
        if (d < R2SQ) idx2[base + cnt++] = i;
    }
    for (; cnt < K_; ++cnt) idx2[base + cnt] = NP1_ - 1;
}

// ---------------------------------------------------------------------------
// Weight fp32 -> f16 conversion with optional zero K-padding.
// ---------------------------------------------------------------------------
__global__ void convert_pad_kernel(const float* __restrict__ src,
                                   _Float16* __restrict__ dst,
                                   int srcRows, int dstRows, int cols)
{
    const int e = blockIdx.x * blockDim.x + threadIdx.x;
    if (e >= dstRows * cols) return;
    const int r = e / cols;
    const int c = e - r * cols;
    dst[e] = (r < srcRows) ? (_Float16)src[r * cols + c] : (_Float16)0.f;
}

// ---------------------------------------------------------------------------
// mlp1: one block (4 waves) per SA1 centroid. 3->64->64->128, max over K.
// ---------------------------------------------------------------------------
#define AS1 136   // LDS activation row stride in halves (68 dwords, conflict-free)

__global__ __launch_bounds__(128) void mlp1_kernel(
    const float* __restrict__ points, const int* __restrict__ idx1,
    const _Float16* __restrict__ w0, const float* __restrict__ b0,
    const _Float16* __restrict__ w1, const float* __restrict__ b1,
    const _Float16* __restrict__ w2, const float* __restrict__ b2,
    float* __restrict__ points3)
{
    __shared__ _Float16 actA[64 * AS1];
    __shared__ _Float16 actB[64 * AS1];
    const int bc  = blockIdx.x;        // b*512 + centroid
    const int b   = bc >> 9;
    const int tid = threadIdx.x;

    if (tid < 64) {
        const int r = tid;
        float x = 0.f, y = 0.f, z = 0.f;
        if (r < K_) {
            const int id = idx1[bc * K_ + r];
            const float* p = points + (size_t)(b * N_ + id) * 3;
            x = p[0]; y = p[1]; z = p[2];
        }
        actA[r * AS1 + 0] = (_Float16)x;
        actA[r * AS1 + 1] = (_Float16)y;
        actA[r * AS1 + 2] = (_Float16)z;
        for (int k = 3; k < 32; ++k) actA[r * AS1 + k] = (_Float16)0.f;
    }
    __syncthreads();
    wave_gemm(actA, AS1, w0, 64, b0, actB, AS1, nullptr, 0, 4, 4, 1, tid);
    __syncthreads();
    wave_gemm(actB, AS1, w1, 64, b1, actA, AS1, nullptr, 0, 4, 4, 2, tid);
    __syncthreads();
    wave_gemm(actA, AS1, w2, 128, b2, actB, AS1, nullptr, 0, 4, 8, 2, tid);
    __syncthreads();
    if (tid < 128) {
        float m = 0.f;
        for (int r = 0; r < K_; ++r) m = fmaxf(m, (float)actB[r * AS1 + tid]);
        points3[(size_t)bc * 128 + tid] = m;
    }
}

// ---------------------------------------------------------------------------
// mlp2: one block per SA2 centroid. 128->128->128->256, fused max over K.
// Emits points4 directly as f16 (feeds the WMMA head).
// ---------------------------------------------------------------------------
__global__ __launch_bounds__(128) void mlp2_kernel(
    const float* __restrict__ points3, const int* __restrict__ idx2,
    const _Float16* __restrict__ w0, const float* __restrict__ b0,
    const _Float16* __restrict__ w1, const float* __restrict__ b1,
    const _Float16* __restrict__ w2, const float* __restrict__ b2,
    _Float16* __restrict__ points4)
{
    __shared__ _Float16 actA[64 * AS1];
    __shared__ _Float16 actB[64 * AS1];
    __shared__ int sidx[64];
    __shared__ int smaxi[256];
    const int bc  = blockIdx.x;        // b*128 + centroid
    const int b   = bc >> 7;
    const int tid = threadIdx.x;

    if (tid < 64) sidx[tid] = (tid < K_) ? idx2[bc * K_ + tid] : 0;
    for (int e = tid; e < 256; e += 128) smaxi[e] = 0;
    __syncthreads();
    for (int e = tid; e < 64 * 128; e += 128) {
        const int r = e >> 7, k = e & 127;
        const float v = (r < K_) ? points3[(size_t)(b * NP1_ + sidx[r]) * 128 + k] : 0.f;
        actA[r * AS1 + k] = (_Float16)v;
    }
    __syncthreads();
    wave_gemm(actA, AS1, w0, 128, b0, actB, AS1, nullptr, 0, 4, 8, 4, tid);
    __syncthreads();
    wave_gemm(actB, AS1, w1, 128, b1, actA, AS1, nullptr, 0, 4, 8, 4, tid);
    __syncthreads();
    wave_gemm(actA, AS1, w2, 256, b2, nullptr, 0, smaxi, K_, 4, 16, 4, tid);
    __syncthreads();
    for (int c = tid; c < 256; c += 128) {
        points4[(size_t)bc * 256 + c] = (_Float16)__int_as_float(smaxi[c]);
    }
}

// ---------------------------------------------------------------------------
// MLP head GEMM, f16 in / f16 out, bias+relu. 64x64 tile per block (4 waves).
// A tile: GLOBAL_LOAD_ASYNC_TO_LDS_B128 (ASYNCcnt).
// W tile: TENSOR_LOAD_TO_LDS via TDM descriptor (TENSORcnt), wave0 issues.
// ---------------------------------------------------------------------------
#define AS3 40   // sA row stride in halves (80 B: 16B-aligned, bank-friendly)

__global__ __launch_bounds__(128) void gemm_head_kernel(
    const _Float16* __restrict__ A, const _Float16* __restrict__ W,
    const float* __restrict__ bias, _Float16* __restrict__ O,
    int M, int Kd, int N, int do_relu)
{
    __shared__ __align__(16) _Float16 sA[64 * AS3];
    __shared__ __align__(16) _Float16 sB[32 * 64];
    const int tid   = threadIdx.x;
    const int nBase = blockIdx.x * 64;
    const int mBase = blockIdx.y * 64;
    const int wave  = tid >> 5;
    const int lane  = tid & 31;
    const int l15   = lane & 15;
    const int h     = lane >> 4;

    v8f acc[4];
#pragma unroll
    for (int t = 0; t < 4; ++t) acc[t] = zero8();

    for (int kb = 0; kb < Kd; kb += 32) {
#if USE_ASYNC
        // A tile: 64 rows x 64 B  -> 256 x 16 B chunks (2 per thread)
#pragma unroll
        for (int c = tid; c < 256; c += 128) {
            const int r = c >> 2, q = c & 3;
            async_copy16(lds_off(&sA[r * AS3]) + q * 16,
                         (const void*)(A + (size_t)(mBase + r) * Kd + kb + q * 8));
        }
#if HAS_TDM
        // W tile: one TDM descriptor moves 32 x 64 f16 (row stride N) into sB
        if (wave == 0) {
            tdm_load_tile_f16(lds_off(sB),
                              (const void*)(W + (size_t)kb * N + nBase),
                              32u, 64u, (unsigned)N);
        }
#else
        // W tile: 32 rows x 128 B -> 256 x 16 B chunks (2 per thread)
#pragma unroll
        for (int c = tid; c < 256; c += 128) {
            const int r = c >> 3, q = c & 7;
            async_copy16(lds_off(&sB[r * 64]) + q * 16,
                         (const void*)(W + (size_t)(kb + r) * N + nBase + q * 8));
        }
#endif
        if (kb + 32 < Kd) {
            __builtin_prefetch(A + (size_t)(mBase + (tid >> 1)) * Kd + kb + 32, 0, 3);
        }
        async_wait0();
#if HAS_TDM
        if (wave == 0) __builtin_amdgcn_s_wait_tensorcnt(0);
#endif
#else
        for (int e = tid; e < 64 * 32; e += 128) {
            const int r = e >> 5, k = e & 31;
            sA[r * AS3 + k] = A[(size_t)(mBase + r) * Kd + kb + k];
        }
        for (int e = tid; e < 32 * 64; e += 128) {
            const int k = e >> 6, n = e & 63;
            sB[e] = W[(size_t)(kb + k) * N + nBase + n];
        }
#endif
        __syncthreads();
        v16h af;
#pragma unroll
        for (int v = 0; v < 8; ++v) {
            const int ka = (v < 4) ? (8 * h + 2 * v) : (16 + 8 * h + 2 * (v - 4));
            af[2 * v]     = sA[(16 * wave + l15) * AS3 + ka];
            af[2 * v + 1] = sA[(16 * wave + l15) * AS3 + ka + 1];
        }
#pragma unroll
        for (int t = 0; t < 4; ++t) {
            v16h bf;
#pragma unroll
            for (int v = 0; v < 8; ++v) {
                const int k2 = (v < 4) ? (8 * h + 2 * v) : (16 + 8 * h + 2 * (v - 4));
                bf[2 * v]     = sB[k2 * 64 + 16 * t + l15];
                bf[2 * v + 1] = sB[(k2 + 1) * 64 + 16 * t + l15];
            }
            acc[t] = __builtin_amdgcn_wmma_f32_16x16x32_f16(
                false, af, false, bf, (short)0, acc[t], false, false);
        }
        __syncthreads();
    }
#pragma unroll
    for (int t = 0; t < 4; ++t) {
        const int ncol = nBase + 16 * t + l15;
        const float bv = bias[ncol];
#pragma unroll
        for (int r = 0; r < 8; ++r) {
            const int Mr = mBase + 16 * wave + r + 8 * h;
            float val = acc[t][r] + bv;
            if (do_relu) val = fmaxf(val, 0.f);
            O[(size_t)Mr * N + ncol] = (_Float16)val;
        }
    }
}

// ---------------------------------------------------------------------------
// Final global max over the 128 centroids (f16 head output) + assembly.
// ---------------------------------------------------------------------------
__global__ void maxrows_kernel(const _Float16* __restrict__ in, float* __restrict__ x)
{
    const int gid = blockIdx.x * blockDim.x + threadIdx.x;
    if (gid >= B_ * 2048) return;
    const int b = gid >> 11, c = gid & 2047;
    const _Float16* p = in + (size_t)b * NP2_ * 2048 + c;
    float m = (float)p[0];
    for (int j = 1; j < NP2_; ++j) m = fmaxf(m, (float)p[(size_t)j * 2048]);
    x[gid] = m;
}

__global__ void finalize_kernel(const float* __restrict__ x,
                                const float* __restrict__ ym,
                                const float* __restrict__ pr,
                                const float* __restrict__ fo,
                                const float* __restrict__ pv,
                                float* __restrict__ out)
{
    const int gid = blockIdx.x * blockDim.x + threadIdx.x;
    const int total = 2 * B_ * 1028;
    if (gid >= total) return;
    const int half = gid / (B_ * 1028);
    const int rem  = gid - half * (B_ * 1028);
    const int b = rem / 1028, c = rem % 1028;
    float v;
    if (half == 0) {
        if (c < 1024)        v = x[b * 2048 + c];
        else if (c == 1024)  v = ym[b];
        else if (c == 1025)  v = pr[b];
        else if (c == 1026)  v = fo[b];
        else                 v = pv[b];
    } else {
        if (c < 1024)        v = x[b * 2048 + 1024 + c];
        else if (c < 1027)   v = 0.1f;
        else                 v = 1.0f;
    }
    out[gid] = v;
}

// ---------------------------------------------------------------------------
// Host launch
// ---------------------------------------------------------------------------
extern "C" void kernel_launch(void* const* d_in, const int* in_sizes, int n_in,
                              void* d_out, int out_size, void* d_ws, size_t ws_size,
                              hipStream_t stream)
{
    (void)in_sizes; (void)n_in; (void)out_size; (void)ws_size;

    const float* points = (const float*)d_in[0];
    const float* young  = (const float*)d_in[1];
    const float* pois   = (const float*)d_in[2];
    const float* force  = (const float*)d_in[3];
    const float* poa    = (const float*)d_in[4];
    // params1
    const float* p1W0 = (const float*)d_in[5];
    const float* p1b0 = (const float*)d_in[6];
    const float* p1W1 = (const float*)d_in[7];
    const float* p1b1 = (const float*)d_in[8];
    const float* p1W2 = (const float*)d_in[9];
    const float* p1b2 = (const float*)d_in[10];
    // params2
    const float* p2W0 = (const float*)d_in[11];
    const float* p2b0 = (const float*)d_in[12];
    const float* p2W1 = (const float*)d_in[13];
    const float* p2b1 = (const float*)d_in[14];
    const float* p2W2 = (const float*)d_in[15];
    const float* p2b2 = (const float*)d_in[16];
    // params3
    const float* p3W0 = (const float*)d_in[17];
    const float* p3b0 = (const float*)d_in[18];
    const float* p3W1 = (const float*)d_in[19];
    const float* p3b1 = (const float*)d_in[20];
    const float* p3W2 = (const float*)d_in[21];
    const float* p3b2 = (const float*)d_in[22];
    const float* p3W3 = (const float*)d_in[23];
    const float* p3b3 = (const float*)d_in[24];

    float* out = (float*)d_out;

    // workspace sub-allocation (256B aligned)
    char* ws = (char*)d_ws;
    size_t off = 0;
    auto alloc = [&](size_t bytes) -> void* {
        void* p = ws + off;
        off = (off + bytes + 255) & ~(size_t)255;
        return p;
    };
    int*      c1_idx  = (int*)alloc((size_t)B_ * NP1_ * 4);
    int*      idx1    = (int*)alloc((size_t)B_ * NP1_ * K_ * 4);
    int*      c2_idx  = (int*)alloc((size_t)B_ * NP2_ * 4);
    int*      idx2    = (int*)alloc((size_t)B_ * NP2_ * K_ * 4);
    float*    points3 = (float*)alloc((size_t)B_ * NP1_ * 128 * 4);
    _Float16* points4 = (_Float16*)alloc((size_t)B_ * NP2_ * 256 * 2);
    _Float16* ping    = (_Float16*)alloc((size_t)2048 * 2048 * 2);
    _Float16* pong    = (_Float16*)alloc((size_t)2048 * 2048 * 2);
    float*    xbuf    = (float*)alloc((size_t)B_ * 2048 * 4);
    _Float16* fw10 = (_Float16*)alloc((size_t)32 * 64 * 2);
    _Float16* fw11 = (_Float16*)alloc((size_t)64 * 64 * 2);
    _Float16* fw12 = (_Float16*)alloc((size_t)64 * 128 * 2);
    _Float16* fw20 = (_Float16*)alloc((size_t)128 * 128 * 2);
    _Float16* fw21 = (_Float16*)alloc((size_t)128 * 128 * 2);
    _Float16* fw22 = (_Float16*)alloc((size_t)128 * 256 * 2);
    _Float16* fw30 = (_Float16*)alloc((size_t)256 * 256 * 2);
    _Float16* fw31 = (_Float16*)alloc((size_t)256 * 512 * 2);
    _Float16* fw32 = (_Float16*)alloc((size_t)512 * 1024 * 2);
    _Float16* fw33 = (_Float16*)alloc((size_t)1024 * 2048 * 2);

    // --- weight conversion (f32 -> f16, zero pad K of first layer to 32) ---
    auto cvt = [&](const float* src, _Float16* dst, int sr, int dr, int cols) {
        int n = dr * cols;
        convert_pad_kernel<<<(n + 255) / 256, 256, 0, stream>>>(src, dst, sr, dr, cols);
    };
    cvt(p1W0, fw10, 3, 32, 64);
    cvt(p1W1, fw11, 64, 64, 64);
    cvt(p1W2, fw12, 64, 64, 128);
    cvt(p2W0, fw20, 128, 128, 128);
    cvt(p2W1, fw21, 128, 128, 128);
    cvt(p2W2, fw22, 128, 128, 256);
    cvt(p3W0, fw30, 256, 256, 256);
    cvt(p3W1, fw31, 256, 256, 512);
    cvt(p3W2, fw32, 512, 512, 1024);
    cvt(p3W3, fw33, 1024, 1024, 2048);

    // --- SA stage 1 ---
    fps1_kernel<<<B_, 1024, 0, stream>>>(points, c1_idx);
    ballq1_kernel<<<(B_ * NP1_ + 255) / 256, 256, 0, stream>>>(points, c1_idx, idx1);
    mlp1_kernel<<<B_ * NP1_, 128, 0, stream>>>(points, idx1,
                                               fw10, p1b0, fw11, p1b1, fw12, p1b2,
                                               points3);

    // --- SA stage 2 ---
    fps2_kernel<<<B_, 512, 0, stream>>>(points3, c2_idx);
    ballq2_kernel<<<(B_ * NP2_ + 255) / 256, 256, 0, stream>>>(points3, c2_idx, idx2);
    mlp2_kernel<<<B_ * NP2_, 128, 0, stream>>>(points3, idx2,
                                               fw20, p2b0, fw21, p2b1, fw22, p2b2,
                                               points4);

    // --- MLP head: 2048 rows, 256->256->512->1024->2048 (relu all layers) ---
    gemm_head_kernel<<<dim3(256 / 64, 2048 / 64), 128, 0, stream>>>(
        points4, fw30, p3b0, ping, 2048, 256, 256, 1);
    gemm_head_kernel<<<dim3(512 / 64, 2048 / 64), 128, 0, stream>>>(
        ping, fw31, p3b1, pong, 2048, 256, 512, 1);
    gemm_head_kernel<<<dim3(1024 / 64, 2048 / 64), 128, 0, stream>>>(
        pong, fw32, p3b2, ping, 2048, 512, 1024, 1);
    gemm_head_kernel<<<dim3(2048 / 64, 2048 / 64), 128, 0, stream>>>(
        ping, fw33, p3b3, pong, 2048, 1024, 2048, 1);

    // --- global max + output assembly ---
    maxrows_kernel<<<(B_ * 2048 + 255) / 256, 256, 0, stream>>>(pong, xbuf);
    finalize_kernel<<<(2 * B_ * 1028 + 255) / 256, 256, 0, stream>>>(
        xbuf, young, pois, force, poa, out);
}
